// Model_39676907885817
// MI455X (gfx1250) — matrix operand order repaired and checked by reading, not compiled, not run on hardware
//
#include <hip/hip_runtime.h>
#include <hip/hip_fp16.h>

// Problem constants (B=4, L=4096, H=16, D=64, dm=1024)
#define TOK     16384          // B*L
#define DM      1024           // H*D
#define ROWS    262144         // B*L*H  (token-head rows, K=64 each)

typedef __attribute__((ext_vector_type(16))) _Float16 v16h;
typedef __attribute__((ext_vector_type(8)))  _Float16 v8h;
typedef __attribute__((ext_vector_type(8)))  float    v8f;
typedef __attribute__((ext_vector_type(4)))  int      v4i;

union V16 { v16h v; v8h h[2]; };

// ---------------------------------------------------------------------------
// CDNA5 async direct-to-LDS copy (GLOBAL_LOAD_ASYNC_TO_LDS_B128, ASYNCcnt).
// Builtin signature (from compiler diagnostic): arg0 = AS(1) v4i* (global),
// arg1 = AS(3) v4i* (LDS), then imm offset + cpol.
// Falls back to a VGPR round-trip copy when the builtin is unavailable
// (host pass / older toolchains).
// ---------------------------------------------------------------------------
#if defined(__gfx1250__) && __has_builtin(__builtin_amdgcn_global_load_async_to_lds_b128)
#define HAVE_ASYNC_LDS 1
#endif

typedef __attribute__((address_space(1))) v4i* gptr_b128;
typedef __attribute__((address_space(3))) v4i* lptr_b128;

__device__ __forceinline__ void copy16_g2l(const _Float16* g, _Float16* l)
{
#ifdef HAVE_ASYNC_LDS
    __builtin_amdgcn_global_load_async_to_lds_b128((gptr_b128)g, (lptr_b128)l, 0, 0);
#else
    *(v8h*)l = *(const v8h*)g;
#endif
}

__device__ __forceinline__ void async_wait0()
{
#ifdef HAVE_ASYNC_LDS
#if __has_builtin(__builtin_amdgcn_s_wait_asynccnt)
    __builtin_amdgcn_s_wait_asynccnt(0);
#else
    asm volatile("s_wait_asynccnt 0" ::: "memory");
#endif
#endif
}

// ---------------------------------------------------------------------------
// Prepass: convert weights to f16 in B-matrix (K-major) order.
// WoT[k*1024+n] = Wo[n][k]; WqT/WkT/WvT[k*64+n] = W[n][k].
// ---------------------------------------------------------------------------
__global__ __launch_bounds__(256) void cvt_weights_kernel(
    const float* __restrict__ Wq, const float* __restrict__ Wk,
    const float* __restrict__ Wv, const float* __restrict__ Wo,
    _Float16* __restrict__ WqT, _Float16* __restrict__ WkT,
    _Float16* __restrict__ WvT, _Float16* __restrict__ WoT)
{
    const int idx = blockIdx.x * blockDim.x + threadIdx.x;
    if (idx < 1024 * 1024) {
        const int k = idx >> 10, n = idx & 1023;
        WoT[idx] = (_Float16)Wo[n * 1024 + k];
    }
    if (idx < 64 * 64) {
        const int k = idx >> 6, n = idx & 63;
        WqT[idx] = (_Float16)Wq[n * 64 + k];
        WkT[idx] = (_Float16)Wk[n * 64 + k];
        WvT[idx] = (_Float16)Wv[n * 64 + k];
    }
}

// ---------------------------------------------------------------------------
// Stage 1: per-head QKV projections + linear-attention gate.
//   rows = token-heads [ROWS x 64]; per wave: 16 rows.
//   q/k/v = A(16x64) @ W^T(64x64) via v_wmma_f32_16x16x32_f16 (2 K-steps).
//   kv[row] = sum_e (k+bk)(v+bv); inner = (q+bq) * kv * sqrt(64).
//   inner stored f16; [ROWS x 64] flat == [16384 x 1024] row-major.
// ---------------------------------------------------------------------------
#define S1_ROWS 128
#define XS 72          // LDS row stride (halves) for x tile  (mult. of 8)
#define WS 72          // LDS row stride (halves) for weights (mult. of 8)

__global__ __launch_bounds__(256) void qkv_gate_kernel(
    const float* __restrict__ x,        // [ROWS,64] f32 (flat view of [B,L,dm])
    const _Float16* __restrict__ WqT, const _Float16* __restrict__ WkT,
    const _Float16* __restrict__ WvT,
    const float* __restrict__ bq, const float* __restrict__ bk,
    const float* __restrict__ bv,
    _Float16* __restrict__ inner)       // [ROWS,64] f16
{
    __shared__ _Float16 xT[S1_ROWS * XS];
    __shared__ _Float16 wT[3][64 * WS];

    const int  tid     = threadIdx.x;
    const long rowBase = (long)blockIdx.x * S1_ROWS;

    // Weight tiles: f16 already -> async direct-to-LDS (8-half chunks).
    {
        const _Float16* wsrc[3] = { WqT, WkT, WvT };
        for (int i = tid; i < 3 * 512; i += 256) {
            const int mat = i >> 9;
            const int rem = i & 511;
            const int k   = rem >> 3;
            const int n8  = (rem & 7) * 8;
            copy16_g2l(&wsrc[mat][k * 64 + n8], &wT[mat][k * WS + n8]);
        }
    }

    // x tile: 128 rows x 64 f32 -> f16 in LDS (needs conversion, VALU path)
    for (int i = tid; i < S1_ROWS * 16; i += 256) {
        const int r  = i >> 4;
        const int c4 = i & 15;
        const float4 f = ((const float4*)(x + (rowBase + r) * 64))[c4];
        _Float16* d = &xT[r * XS + c4 * 4];
        d[0] = (_Float16)f.x; d[1] = (_Float16)f.y;
        d[2] = (_Float16)f.z; d[3] = (_Float16)f.w;
    }
    async_wait0();
    __syncthreads();

    const int wave    = tid >> 5;
    const int lane    = tid & 31;
    const int mrow    = lane & 15;   // C/B fragment: col-in-frag; A: row
    const int hi      = lane >> 4;   // lane half
    const int waveRow = wave * 16;

    // A fragments (16x32 f16): lane<16 holds K {0..7,16..23}, lane>=16 {8..15,24..31}
    V16 a0, a1;
    {
        const int m  = waveRow + mrow;
        const int kb = hi * 8;
        a0.h[0] = *(const v8h*)&xT[m * XS + kb];
        a0.h[1] = *(const v8h*)&xT[m * XS + kb + 16];
        a1.h[0] = *(const v8h*)&xT[m * XS + 32 + kb];
        a1.h[1] = *(const v8h*)&xT[m * XS + 32 + kb + 16];
    }

    v8f qacc[4];
    v8f prod = {};

#pragma unroll
    for (int f = 0; f < 4; ++f) {
        const int n0 = f * 16;
        v8f q = {}, k = {}, v = {};
#pragma unroll
        for (int ks = 0; ks < 2; ++ks) {
            V16 a = ks ? a1 : a0;
            const int nc = n0 + mrow;  // B frag: lane%16 == N col, K in table order
            V16 b;
#pragma unroll
            for (int s = 0; s < 16; ++s) b.v[s] = wT[0][(ks * 32 + hi * 8 + (s & 7) + (s >> 3) * 16) * WS + nc];
            q = __builtin_amdgcn_wmma_f32_16x16x32_f16(false, a.v, false, b.v,
                                                       (short)0, q, false, false);
#pragma unroll
            for (int s = 0; s < 16; ++s) b.v[s] = wT[1][(ks * 32 + hi * 8 + (s & 7) + (s >> 3) * 16) * WS + nc];
            k = __builtin_amdgcn_wmma_f32_16x16x32_f16(false, a.v, false, b.v,
                                                       (short)0, k, false, false);
#pragma unroll
            for (int s = 0; s < 16; ++s) b.v[s] = wT[2][(ks * 32 + hi * 8 + (s & 7) + (s >> 3) * 16) * WS + nc];
            v = __builtin_amdgcn_wmma_f32_16x16x32_f16(false, a.v, false, b.v,
                                                       (short)0, v, false, false);
        }
        const float bqv = bq[n0 + mrow];
        const float bkv = bk[n0 + mrow];
        const float bvv = bv[n0 + mrow];
#pragma unroll
        for (int j = 0; j < 8; ++j) {
            const float kk = k[j] + bkv;
            const float vv = v[j] + bvv;
            prod[j] += kk * vv;
            q[j] += bqv;
        }
        qacc[f] = q;
    }

    // Reduce kv across the 16-lane half (matches C-fragment row mapping),
    // fold in scale sqrt(D) = 8.
#pragma unroll
    for (int j = 0; j < 8; ++j) {
        float s = prod[j];
        s += __shfl_xor(s, 1, 32);
        s += __shfl_xor(s, 2, 32);
        s += __shfl_xor(s, 4, 32);
        s += __shfl_xor(s, 8, 32);
        prod[j] = s * 8.0f;
    }

#pragma unroll
    for (int f = 0; f < 4; ++f) {
#pragma unroll
        for (int j = 0; j < 8; ++j) {
            const long row = rowBase + waveRow + j + hi * 8;
            const int  col = f * 16 + mrow;
            inner[row * 64 + col] = (_Float16)(qacc[f][j] * prod[j]);
        }
    }
}

// ---------------------------------------------------------------------------
// Stage 2: out[16384,1024] = inner @ Wo^T + bo  (f16 WMMA, f32 accumulate)
//   Block tile 128x128, BK=32, 8 waves -> each wave 32x64 (2x4 C frags).
//   Double-buffered LDS, filled with GLOBAL_LOAD_ASYNC_TO_LDS_B128;
//   global_prefetch_b8 warms the K-panel after next.
// ---------------------------------------------------------------------------
#define AS 40          // A tile LDS stride (halves)
#define BS 136         // B tile LDS stride (halves)

__global__ __launch_bounds__(256) void out_proj_kernel(
    const _Float16* __restrict__ inner,  // [16384,1024]
    const _Float16* __restrict__ WoT,    // [1024,1024] K-major
    const float* __restrict__ bo,        // [1024]
    float* __restrict__ out)             // [16384,1024]
{
    __shared__ _Float16 At[2][128 * AS];
    __shared__ _Float16 Bt[2][32 * BS];

    const int  tid = threadIdx.x;
    const long m0  = (long)blockIdx.x * 128;
    const int  n0  = blockIdx.y * 128;

    auto loadA = [&](int buf, int k0) {
        for (int i = tid; i < 512; i += 256) {              // 128 rows x 4 octets
            const int r = i >> 2;
            const int c = (i & 3) * 8;
            copy16_g2l(&inner[(m0 + r) * 1024 + k0 + c], &At[buf][r * AS + c]);
        }
    };
    auto loadB = [&](int buf, int k0) {
        for (int i = tid; i < 512; i += 256) {              // 32 rows x 16 octets
            const int r = i >> 4;
            const int c = (i & 15) * 8;
            copy16_g2l(&WoT[(long)(k0 + r) * 1024 + n0 + c], &Bt[buf][r * BS + c]);
        }
    };

    const int wave = tid >> 5, lane = tid & 31;
    const int wm   = (wave & 3) * 32;   // wave row base in tile
    const int wn   = (wave >> 2) * 64;  // wave col base in tile
    const int mrow = lane & 15, hi = lane >> 4;

    v8f acc[2][4] = {};

    loadA(0, 0);
    loadB(0, 0);
    async_wait0();
    __syncthreads();

    for (int kt = 0; kt < 32; ++kt) {
        const int cur = kt & 1, nxt = cur ^ 1;
        if (kt < 31) {
            loadA(nxt, (kt + 1) * 32);
            loadB(nxt, (kt + 1) * 32);
        }
        if (kt < 30) {   // warm L2/WGP$ for the K-panel after next
            __builtin_prefetch(&inner[(m0 + (tid & 127)) * 1024 + (kt + 2) * 32], 0, 0);
            __builtin_prefetch(&WoT[(long)((kt + 2) * 32 + (tid & 31)) * 1024 + n0], 0, 0);
        }

        V16 a[2];
#pragma unroll
        for (int i = 0; i < 2; ++i) {
            const int m  = wm + i * 16 + mrow;
            const int kb = hi * 8;
            a[i].h[0] = *(const v8h*)&At[cur][m * AS + kb];
            a[i].h[1] = *(const v8h*)&At[cur][m * AS + kb + 16];
        }
#pragma unroll
        for (int f = 0; f < 4; ++f) {
            V16 b;
            const int bc = wn + f * 16 + mrow;
#pragma unroll
            for (int s = 0; s < 16; ++s) b.v[s] = Bt[cur][(hi * 8 + (s & 7) + (s >> 3) * 16) * BS + bc];
            acc[0][f] = __builtin_amdgcn_wmma_f32_16x16x32_f16(
                false, a[0].v, false, b.v, (short)0, acc[0][f], false, false);
            acc[1][f] = __builtin_amdgcn_wmma_f32_16x16x32_f16(
                false, a[1].v, false, b.v, (short)0, acc[1][f], false, false);
        }
        // Wave's own async fills for `nxt` must land before the barrier so the
        // whole workgroup can read them next iteration.
        async_wait0();
        __syncthreads();
    }

#pragma unroll
    for (int i = 0; i < 2; ++i) {
#pragma unroll
        for (int f = 0; f < 4; ++f) {
            const int   col  = n0 + wn + f * 16 + mrow;
            const float bias = bo[col];
#pragma unroll
            for (int j = 0; j < 8; ++j) {
                const long row = m0 + wm + i * 16 + j + hi * 8;
                out[row * 1024 + col] = acc[i][f][j] + bias;
            }
        }
    }
}

// ---------------------------------------------------------------------------
extern "C" void kernel_launch(void* const* d_in, const int* in_sizes, int n_in,
                              void* d_out, int out_size, void* d_ws, size_t ws_size,
                              hipStream_t stream)
{
    const float* x  = (const float*)d_in[0];
    const float* Wq = (const float*)d_in[1];
    const float* bq = (const float*)d_in[2];
    const float* Wk = (const float*)d_in[3];
    const float* bk = (const float*)d_in[4];
    const float* Wv = (const float*)d_in[5];
    const float* bv = (const float*)d_in[6];
    const float* Wo = (const float*)d_in[7];
    const float* bo = (const float*)d_in[8];
    float* out = (float*)d_out;

    char* ws = (char*)d_ws;
    _Float16* inner = (_Float16*)ws;  ws += (size_t)ROWS * 64 * 2;      // 32 MB
    _Float16* WoT   = (_Float16*)ws;  ws += (size_t)1024 * 1024 * 2;    //  2 MB
    _Float16* WqT   = (_Float16*)ws;  ws += 64 * 64 * 2;
    _Float16* WkT   = (_Float16*)ws;  ws += 64 * 64 * 2;
    _Float16* WvT   = (_Float16*)ws;  ws += 64 * 64 * 2;

    cvt_weights_kernel<<<(1024 * 1024 + 255) / 256, 256, 0, stream>>>(
        Wq, Wk, Wv, Wo, WqT, WkT, WvT, WoT);

    qkv_gate_kernel<<<ROWS / S1_ROWS, 256, 0, stream>>>(
        x, WqT, WkT, WvT, bq, bk, bv, inner);

    dim3 g2(TOK / 128, DM / 128);
    out_proj_kernel<<<g2, 256, 0, stream>>>(inner, WoT, bo, out);
}
